// MultiHeadSelfAttentionLayer_42477226557531
// MI455X (gfx1250) — matrix operand order repaired and checked by reading, not compiled, not run on hardware
//
#include <hip/hip_runtime.h>
#include <hip/hip_bf16.h>
#include <math.h>

// ---------------------------------------------------------------------------
// MI455X (gfx1250) implementation.
// All matmul work goes through v_wmma_f32_16x16x32_bf16 (wave32 WMMA).
// Weights are pre-transposed at convert time so BOTH WMMA operand fragments
// are built from contiguous 16B global loads -> no LDS in any GEMM inner loop.
// Sizes fixed by the reference: N=4, L=2048, E=HID=1024, HEADS=16, d=64.
// ---------------------------------------------------------------------------

typedef __attribute__((ext_vector_type(16))) __bf16 v16bf;
typedef __attribute__((ext_vector_type(8)))  __bf16 v8bf;
typedef __attribute__((ext_vector_type(8)))  float  v8f;

#define NB   4
#define LSEQ 2048
#define EMB  1024
#define HID  1024
#define HEADS 16
#define MROWS (NB*LSEQ)          // 8192

static __device__ __forceinline__ __bf16 f2bf(float f) {
    union { float f; unsigned int u; } v; v.f = f;
    unsigned int u = v.u;
    unsigned short r;
    if ((u & 0x7fffffffu) > 0x7f800000u) {           // NaN
        r = 0x7fc0;
    } else {                                          // round-to-nearest-even
        r = (unsigned short)((u + 0x7fffu + ((u >> 16) & 1u)) >> 16);
    }
    return __builtin_bit_cast(__bf16, r);
}

// ------------------------------- fp32 -> bf16 ------------------------------
__global__ void k_cvt_bf16(const float* __restrict__ src, __bf16* __restrict__ dst, int n) {
    int i = blockIdx.x * blockDim.x + threadIdx.x;
    if (i < n) dst[i] = f2bf(src[i]);
}

// ------------------- fp32 W[k][n] -> bf16 Wt[n][k] (1024x1024) -------------
// Tiled LDS transpose: coalesced read and coalesced write.
__global__ void k_cvt_w_t(const float* __restrict__ src, __bf16* __restrict__ dst) {
    __shared__ float tile[32][33];
    const int bk = blockIdx.y * 32;                   // k tile base (src row)
    const int bn = blockIdx.x * 32;                   // n tile base (src col)
    const int tx = threadIdx.x & 31;
    const int ty = threadIdx.x >> 5;                  // 0..7
    for (int r = 0; r < 32; r += 8)
        tile[ty + r][tx] = src[(size_t)(bk + ty + r) * 1024 + bn + tx];
    __syncthreads();
    for (int r = 0; r < 32; r += 8)                   // dst[n][k] = src[k][n]
        dst[(size_t)(bn + ty + r) * 1024 + bk + tx] = f2bf(tile[tx][ty + r]);
}

// ------------------------- tiled WMMA bf16 GEMM ----------------------------
// C[M,1024] = (A[M,1024] @ Bt[1024,1024]^T + bias) * scale,  Bt is N x K.
// Block tile 128x128; 256 threads = 8 waves; each wave: 2x4 tiles of 16x16.
// No LDS: both fragment types are contiguous 16B global loads (L1/L2 cached).
template <bool F32OUT>
__global__ void k_gemm_bf16(const __bf16* __restrict__ A,
                            const __bf16* __restrict__ Bt,
                            const float*  __restrict__ bias,
                            void*         __restrict__ Cout,
                            float scale) {
    const int wave = threadIdx.x >> 5;
    const int lane = threadIdx.x & 31;
    const int mrow = wave >> 1;                       // 0..3
    const int ncol = wave & 1;                        // 0..1
    const int blockM = blockIdx.y * 128;
    const int blockN = blockIdx.x * 128;

    v8f acc[2][4];
    for (int i = 0; i < 2; ++i)
        for (int t = 0; t < 4; ++t)
            for (int j = 0; j < 8; ++j) acc[i][t][j] = 0.0f;

    const int am  = lane & 15;
    const int ako = (lane >> 4) * 8;                  // A fragment k sub-offset
    const int bn  = lane & 15;
    const int bko = (lane >> 4) * 8;                  // B fragment k sub-offset

    // per-lane base pointers (k advances by 32 each step)
    const __bf16* ap0 = A  + (size_t)(blockM + mrow * 32 +      am) * 1024 + ako;
    const __bf16* ap1 = A  + (size_t)(blockM + mrow * 32 + 16 + am) * 1024 + ako;
    const __bf16* bp0 = Bt + (size_t)(blockN + ncol * 64 +      bn) * 1024 + bko;
    const __bf16* bp1 = Bt + (size_t)(blockN + ncol * 64 + 16 + bn) * 1024 + bko;
    const __bf16* bp2 = Bt + (size_t)(blockN + ncol * 64 + 32 + bn) * 1024 + bko;
    const __bf16* bp3 = Bt + (size_t)(blockN + ncol * 64 + 48 + bn) * 1024 + bko;

    for (int k0 = 0; k0 < 1024; k0 += 32) {
        // A fragments: two 16x32 tiles
        v16bf afr[2];
        {
            v8bf lo0 = *(const v8bf*)(ap0 + k0), hi0 = *(const v8bf*)(ap0 + k0 + 16);
            v8bf lo1 = *(const v8bf*)(ap1 + k0), hi1 = *(const v8bf*)(ap1 + k0 + 16);
            for (int j = 0; j < 8; ++j) {
                afr[0][j] = lo0[j]; afr[0][j + 8] = hi0[j];
                afr[1][j] = lo1[j]; afr[1][j + 8] = hi1[j];
            }
        }
        // B fragments: four 32x16 tiles, contiguous from transposed weights
        v16bf bfr[4];
        {
            const __bf16* bp[4] = { bp0 + k0, bp1 + k0, bp2 + k0, bp3 + k0 };
            for (int t = 0; t < 4; ++t) {
                v8bf lo = *(const v8bf*)bp[t];
                v8bf hi = *(const v8bf*)(bp[t] + 16);
                for (int j = 0; j < 8; ++j) { bfr[t][j] = lo[j]; bfr[t][j + 8] = hi[j]; }
            }
        }
        for (int i = 0; i < 2; ++i)
            for (int t = 0; t < 4; ++t)
                acc[i][t] = __builtin_amdgcn_wmma_f32_16x16x32_bf16(
                    false, afr[i], false, bfr[t], (short)0, acc[i][t], false, false);
    }

    // epilogue: +bias, *scale (C layout: col=lane%16, row=j+8*(lane>>4))
    const int cn = lane & 15;
    const int ro = (lane >> 4) * 8;
    for (int i = 0; i < 2; ++i) {
        for (int t = 0; t < 4; ++t) {
            const int gcol = blockN + ncol * 64 + t * 16 + cn;
            const float bv = bias[gcol];
            const int grow0 = blockM + mrow * 32 + i * 16 + ro;
            for (int j = 0; j < 8; ++j) {
                const float v = (acc[i][t][j] + bv) * scale;
                const size_t off = (size_t)(grow0 + j) * 1024 + gcol;
                if (F32OUT) ((float*)Cout)[off] = v;
                else        ((__bf16*)Cout)[off] = f2bf(v);
            }
        }
    }
}

// ---------------- streamed scores + online softmax + diagonal --------------
// Per (n, head e): S = Qh(2048x64) @ Kh(2048x64)^T, 1/sqrt(HID) pre-folded.
// For each column b: running max/sum over rows a (online softmax), then
// diag[b] = exp(S[b,b]-max_b)/sum_b.  Grid: 1024 blocks x 256 threads;
// each wave owns 16 columns; its B fragments are loop-invariant.
__global__ void k_diag_softmax(const __bf16* __restrict__ Qb,
                               const __bf16* __restrict__ Kb,
                               float* __restrict__ diag) {
    const int blk = blockIdx.x;
    const int cb  = blk & 15;                         // column block (128 cols)
    const int e   = (blk >> 4) & 15;
    const int n   = blk >> 8;
    const int wave = threadIdx.x >> 5;
    const int lane = threadIdx.x & 31;

    const size_t base = (size_t)(n * LSEQ + e * 128) * 1024;  // head block start
    const __bf16* Qh = Qb + base;                     // (2048,64) row-major
    const __bf16* Kh = Kb + base;
    const int btile = cb * 128 + wave * 16;

    // loop-invariant B fragments (K^T): lane -> col btile+(lane&15), k chunks (lane>>4)*8 and +16
    const int bn  = lane & 15;
    const int bko = (lane >> 4) * 8;
    v16bf bf0, bf1;
    {
        const __bf16* kp = Kh + (size_t)(btile + bn) * 64 + bko;
        v8bf a0 = *(const v8bf*)kp,        a1 = *(const v8bf*)(kp + 16);
        v8bf c0 = *(const v8bf*)(kp + 32), c1 = *(const v8bf*)(kp + 48);
        for (int j = 0; j < 8; ++j) {
            bf0[j] = a0[j]; bf0[j + 8] = a1[j];
            bf1[j] = c0[j]; bf1[j + 8] = c1[j];
        }
    }

    float runM = -INFINITY, runS = 0.0f;
    const int am  = lane & 15;
    const int ako = (lane >> 4) * 8;
    for (int at = 0; at < 128; ++at) {                // stream all 2048 rows
        const __bf16* qp = Qh + (size_t)(at * 16 + am) * 64 + ako;
        v8bf l0 = *(const v8bf*)qp,        h0 = *(const v8bf*)(qp + 16);
        v8bf l1 = *(const v8bf*)(qp + 32), h1 = *(const v8bf*)(qp + 48);
        v16bf a0, a1;
        for (int j = 0; j < 8; ++j) {
            a0[j] = l0[j]; a0[j + 8] = h0[j];
            a1[j] = l1[j]; a1[j + 8] = h1[j];
        }
        v8f c;
        for (int j = 0; j < 8; ++j) c[j] = 0.0f;
        c = __builtin_amdgcn_wmma_f32_16x16x32_bf16(false, a0, false, bf0, (short)0, c, false, false);
        c = __builtin_amdgcn_wmma_f32_16x16x32_bf16(false, a1, false, bf1, (short)0, c, false, false);

        float tm = c[0];
        for (int j = 1; j < 8; ++j) tm = fmaxf(tm, c[j]);
        const float nM = fmaxf(runM, tm);
        const float corr = __expf(runM - nM);         // 0 on first iteration
        float s = 0.0f;
        for (int j = 0; j < 8; ++j) s += __expf(c[j] - nM);
        runS = runS * corr + s;
        runM = nM;
    }

    // combine the two half-columns (lane L holds row-offsets 0..7, L+16 holds 8..15)
    const float oM = __shfl_xor(runM, 16, 32);
    const float oS = __shfl_xor(runS, 16, 32);
    const float M  = fmaxf(runM, oM);
    const float S  = runS * __expf(runM - M) + oS * __expf(oM - M);

    if (lane < 16) {
        const int b = btile + lane;
        const __bf16* qp = Qh + (size_t)b * 64;
        const __bf16* kp = Kh + (size_t)b * 64;
        float d = 0.0f;
        for (int x = 0; x < 64; ++x) d += (float)qp[x] * (float)kp[x];   // S[b,b]
        diag[(size_t)(n * HEADS + e) * LSEQ + b] = __expf(d - M) / S;
    }
}

// --------------------------- A2 = diag(.) * V ------------------------------
__global__ void k_scale_v(const __bf16* __restrict__ Vb,
                          const float* __restrict__ diag,
                          __bf16* __restrict__ Ab) {
    const size_t i = (size_t)blockIdx.x * blockDim.x + threadIdx.x;
    const int h   = (int)(i & 1023);
    const int row = (int)(i >> 10);
    const int l   = row & (LSEQ - 1);
    const int n   = row >> 11;
    const int e   = l >> 7;
    const int ii  = ((l & 127) << 4) | (h >> 6);      // plain-reshape index math
    const float dg = diag[(size_t)((n * HEADS + e) << 11) + ii];
    Ab[i] = f2bf(dg * (float)Vb[i]);
}

// ---------------------------------------------------------------------------
extern "C" void kernel_launch(void* const* d_in, const int* in_sizes, int n_in,
                              void* d_out, int out_size, void* d_ws, size_t ws_size,
                              hipStream_t stream) {
    const float* X  = (const float*)d_in[0];
    const float* Wq = (const float*)d_in[1];
    const float* bq = (const float*)d_in[2];
    const float* Wk = (const float*)d_in[3];
    const float* bk = (const float*)d_in[4];
    const float* Wv = (const float*)d_in[5];
    const float* bv = (const float*)d_in[6];
    const float* Wo = (const float*)d_in[7];
    const float* bo = (const float*)d_in[8];

    char* ws = (char*)d_ws;
    const size_t XN = (size_t)MROWS * EMB;            // 8,388,608 elements
    const size_t WN = (size_t)EMB * HID;              // 1,048,576 elements
    size_t off = 0;
    __bf16* Xb  = (__bf16*)(ws + off); off += XN * 2; // 16 MB
    __bf16* Wqt = (__bf16*)(ws + off); off += WN * 2; // transposed bf16 weights
    __bf16* Wkt = (__bf16*)(ws + off); off += WN * 2;
    __bf16* Wvt = (__bf16*)(ws + off); off += WN * 2;
    __bf16* Wot = (__bf16*)(ws + off); off += WN * 2;
    __bf16* Qb  = (__bf16*)(ws + off); off += XN * 2;
    __bf16* Kb  = (__bf16*)(ws + off); off += XN * 2;
    __bf16* Vb  = (__bf16*)(ws + off); off += XN * 2;
    __bf16* Ab  = (__bf16*)(ws + off); off += XN * 2;
    float*  dg  = (float*)(ws + off);  off += (size_t)NB * HEADS * LSEQ * 4;

    // 1. convert X to bf16; convert+transpose weights to bf16 Wt[n][k]
    k_cvt_bf16<<<(int)((XN + 255) / 256), 256, 0, stream>>>(X, Xb, (int)XN);
    dim3 gt(32, 32);
    k_cvt_w_t<<<gt, 256, 0, stream>>>(Wq, Wqt);
    k_cvt_w_t<<<gt, 256, 0, stream>>>(Wk, Wkt);
    k_cvt_w_t<<<gt, 256, 0, stream>>>(Wv, Wvt);
    k_cvt_w_t<<<gt, 256, 0, stream>>>(Wo, Wot);

    // 2. Q/K/V projections (WMMA GEMMs), scale=1/sqrt(HID)=1/32 baked into Q,K
    dim3 gg(HID / 128, MROWS / 128);                  // (8, 64)
    k_gemm_bf16<false><<<gg, 256, 0, stream>>>(Xb, Wqt, bq, Qb, 0.03125f);
    k_gemm_bf16<false><<<gg, 256, 0, stream>>>(Xb, Wkt, bk, Kb, 0.03125f);
    k_gemm_bf16<false><<<gg, 256, 0, stream>>>(Xb, Wvt, bv, Vb, 1.0f);

    // 3. streamed scores + column softmax stats + diagonal
    k_diag_softmax<<<NB * HEADS * (LSEQ / 128), 256, 0, stream>>>(Qb, Kb, dg);

    // 4. apply diagonal attention weight to V
    k_scale_v<<<(int)(XN / 256), 256, 0, stream>>>(Vb, dg, Ab);

    // 5. output projection -> f32 d_out
    k_gemm_bf16<true><<<gg, 256, 0, stream>>>(Ab, Wot, bo, d_out, 1.0f);
}
